// CongestionGNN_38122129719954
// MI455X (gfx1250) — compile-verified, matrix operand order checked by reading, not run on hardware
//
#include <hip/hip_runtime.h>

typedef __attribute__((ext_vector_type(16))) _Float16 v16h;
typedef __attribute__((ext_vector_type(8)))  float    v8f;

// ---------------------------------------------------------------- degree
__global__ void deg_kernel(const long long* __restrict__ dst, float* __restrict__ deg, int E) {
  int e = blockIdx.x * blockDim.x + threadIdx.x;
  if (e < E) atomicAdd(&deg[(int)dst[e]], 1.0f);
}

// ------------------------------------------------- edge gather/scatter-add
// 16 threads per edge, 4 features each: gathers are coalesced b128 loads
// (lanes 0-15 cover one edge's 64 floats), edge-index loads amortized 4x
// vs. one-feature-per-thread. Atomic adds are f32, L2-resident.
__global__ void scatter_kernel(const float* __restrict__ in,
                               const long long* __restrict__ src,
                               const long long* __restrict__ dst,
                               float* __restrict__ agg, int E) {
  long long tid = (long long)blockIdx.x * blockDim.x + threadIdx.x;
  if (tid >= (long long)E * 16) return;
  int e = (int)(tid >> 4);
  int f = ((int)tid & 15) * 4;
  int s = (int)src[e], d = (int)dst[e];
  float4 v = *(const float4*)(in + (size_t)s * 64 + f);
  float* p = agg + (size_t)d * 64 + f;
  atomicAdd(p + 0, v.x);
  atomicAdd(p + 1, v.y);
  atomicAdd(p + 2, v.z);
  atomicAdd(p + 3, v.w);
}

// ---------------------------------------- pack [Wl | Wr] into WMMA B frags
// B fragment (32x16, f16): lane = n + 16*hi holds column n's K values,
// element (vgpr jj, half h): kloc = jj<4 ? 8*hi+2*jj+h : 16+8*hi+2*(jj-4)+h
__global__ void prep_wfrag(const float* __restrict__ Wl, const float* __restrict__ Wr,
                           v16h* __restrict__ frag, int ctiles) {
  int idx = blockIdx.x * blockDim.x + threadIdx.x;
  int lane = idx & 31;
  int t = (idx >> 5) & 3;      // k-tile (K=128 total, 4 tiles of 32)
  int c = idx >> 7;            // column tile
  if (c >= ctiles) return;
  int nn = lane & 15, hi = lane >> 4;
  int col = c * 16 + nn;
  v16h v;
#pragma unroll
  for (int jj = 0; jj < 8; ++jj) {
#pragma unroll
    for (int h = 0; h < 2; ++h) {
      int kloc = (jj < 4) ? (8 * hi + 2 * jj + h) : (16 + 8 * hi + 2 * (jj - 4) + h);
      int k = t * 32 + kloc;
      float w = (k < 64) ? Wl[col * 64 + k] : Wr[col * 64 + (k - 64)];
      v[2 * jj + h] = (_Float16)w;
    }
  }
  frag[(c * 4 + t) * 32 + lane] = v;
}

// ---------------------------------------- A fragment from f32 row (scaled)
// p points at row_base + ktile*32 + 8*hi ; pairs -> global_load_b64
__device__ __forceinline__ v16h load_a(const float* __restrict__ p, float s) {
  v16h a;
#pragma unroll
  for (int jj = 0; jj < 4; ++jj) {
    float2 q = *(const float2*)(p + 2 * jj);
    a[2 * jj]     = (_Float16)(q.x * s);
    a[2 * jj + 1] = (_Float16)(q.y * s);
  }
#pragma unroll
  for (int jj = 0; jj < 4; ++jj) {
    float2 q = *(const float2*)(p + 16 + 2 * jj);
    a[8 + 2 * jj]     = (_Float16)(q.x * s);
    a[8 + 2 * jj + 1] = (_Float16)(q.y * s);
  }
  return a;
}

// ---------------- fused SAGE layer GEMM: out = relu([mean|x] @ Wcat + b)
// block = 256 threads = 8 waves; each wave: 16 rows x DOUT cols.
template <int DOUT>
__global__ void sage_gemm(const float* __restrict__ agg, const float* __restrict__ deg,
                          const float* __restrict__ xin, const float* __restrict__ bias,
                          const v16h* __restrict__ wfrag, float* __restrict__ out, int n) {
  constexpr int CT = DOUT / 16;
  int lane = threadIdx.x & 31;
  int wave = threadIdx.x >> 5;
  int m0 = blockIdx.x * 128 + wave * 16;
  int hi = lane >> 4;
  int row = m0 + (lane & 15);
  int rowc = row < n ? row : n - 1;     // clamp loads; EXEC stays all-ones
  float inv = 1.0f / fmaxf(deg[rowc], 1.0f);

  v8f acc[CT] = {};

#pragma unroll
  for (int t = 0; t < 4; ++t) {
    const float* src = (t < 2) ? agg : xin;   // k<64 = mean side, else root x
    int tb = ((t < 2) ? t : t - 2) * 32;
    float sc = (t < 2) ? inv : 1.0f;
    v16h a = load_a(src + (size_t)rowc * 64 + tb + 8 * hi, sc);
#pragma unroll
    for (int c = 0; c < CT; ++c) {
      v16h b = wfrag[(c * 4 + t) * 32 + lane];
      acc[c] = __builtin_amdgcn_wmma_f32_16x16x32_f16(
          /*neg_a=*/false, a, /*neg_b=*/false, b,
          /*c_mod=*/(short)0, acc[c], /*reuse_a=*/false, /*reuse_b=*/false);
    }
  }

  // C/D layout: lane (0-15|16-31) = col, VGPR r -> row m0 + 8*hi + r
  int rb = m0 + 8 * hi;
  int ncol = lane & 15;
#pragma unroll
  for (int c = 0; c < CT; ++c) {
    int col = c * 16 + ncol;
    float bb = bias[col];
#pragma unroll
    for (int r = 0; r < 8; ++r) {
      int rr = rb + r;
      if (rr < n) {
        float v = acc[c][r] + bb;
        out[(size_t)rr * DOUT + col] = v > 0.f ? v : 0.f;
      }
    }
  }
}

// ------------------------------------------------- final linear regression
__global__ void reg_kernel(const float* __restrict__ h, const float* __restrict__ Wreg,
                           const float* __restrict__ breg, float* __restrict__ out, int n) {
  int i = blockIdx.x * blockDim.x + threadIdx.x;
  if (i >= n) return;
  float s = breg[0];
#pragma unroll
  for (int f = 0; f < 32; f += 4) {
    float4 hv = *(const float4*)(h + (size_t)i * 32 + f);
    float4 wv = *(const float4*)(Wreg + f);   // uniform -> scalar loads
    s += hv.x * wv.x + hv.y * wv.y + hv.z * wv.z + hv.w * wv.w;
  }
  out[i] = s;
}

extern "C" void kernel_launch(void* const* d_in, const int* in_sizes, int n_in,
                              void* d_out, int out_size, void* d_ws, size_t ws_size,
                              hipStream_t stream) {
  (void)n_in; (void)out_size; (void)ws_size;
  const float*     x    = (const float*)d_in[0];
  const long long* ei   = (const long long*)d_in[1];   // int64 (2,E)
  const float* W1l = (const float*)d_in[2];
  const float* b1  = (const float*)d_in[3];
  const float* W1r = (const float*)d_in[4];
  const float* W2l = (const float*)d_in[5];
  const float* b2  = (const float*)d_in[6];
  const float* W2r = (const float*)d_in[7];
  const float* W3l = (const float*)d_in[8];
  const float* b3  = (const float*)d_in[9];
  const float* W3r = (const float*)d_in[10];
  const float* Wreg = (const float*)d_in[11];
  const float* breg = (const float*)d_in[12];

  const int N = in_sizes[0] / 64;   // 100000
  const int E = in_sizes[1] / 2;    // 1600000
  const long long* src = ei;
  const long long* dst = ei + E;

  // carve workspace
  char* ws = (char*)d_ws;
  size_t off = 0;
  auto carve = [&](size_t bytes) -> char* {
    char* p = ws + off;
    off += (bytes + 255) & ~(size_t)255;
    return p;
  };
  float* deg = (float*)carve((size_t)N * 4);
  float* agg = (float*)carve((size_t)N * 64 * 4);
  float* hA  = (float*)carve((size_t)N * 64 * 4);
  float* hB  = (float*)carve((size_t)N * 64 * 4);
  v16h*  f1  = (v16h*)carve(16 * 32 * 32);
  v16h*  f2  = (v16h*)carve(16 * 32 * 32);
  v16h*  f3  = (v16h*)carve(8 * 32 * 32);

  // degree (once) + weight fragment packing (once per launch)
  hipMemsetAsync(deg, 0, (size_t)N * 4, stream);
  deg_kernel<<<(E + 255) / 256, 256, 0, stream>>>(dst, deg, E);
  prep_wfrag<<<2, 256, 0, stream>>>(W1l, W1r, f1, 4);
  prep_wfrag<<<2, 256, 0, stream>>>(W2l, W2r, f2, 4);
  prep_wfrag<<<1, 256, 0, stream>>>(W3l, W3r, f3, 2);

  long long sthreads = (long long)E * 16;
  int sblocks = (int)((sthreads + 255) / 256);
  int gblocks = (N + 127) / 128;

  // layer 1
  hipMemsetAsync(agg, 0, (size_t)N * 64 * 4, stream);
  scatter_kernel<<<sblocks, 256, 0, stream>>>(x, src, dst, agg, E);
  sage_gemm<64><<<gblocks, 256, 0, stream>>>(agg, deg, x, b1, f1, hA, N);

  // layer 2
  hipMemsetAsync(agg, 0, (size_t)N * 64 * 4, stream);
  scatter_kernel<<<sblocks, 256, 0, stream>>>(hA, src, dst, agg, E);
  sage_gemm<64><<<gblocks, 256, 0, stream>>>(agg, deg, hA, b2, f2, hB, N);

  // layer 3 (dout = 32), reuse hA for output
  hipMemsetAsync(agg, 0, (size_t)N * 64 * 4, stream);
  scatter_kernel<<<sblocks, 256, 0, stream>>>(hB, src, dst, agg, E);
  sage_gemm<32><<<gblocks, 256, 0, stream>>>(agg, deg, hB, b3, f3, hA, N);

  // regression head
  reg_kernel<<<(N + 255) / 256, 256, 0, stream>>>(hA, Wreg, breg, (float*)d_out, N);
}
